// LocPreNormConformerBlock_40278203302025
// MI455X (gfx1250) — matrix operand order repaired and checked by reading, not compiled, not run on hardware
//
#include <hip/hip_runtime.h>
#include <hip/hip_bf16.h>

// ---------------------------------------------------------------------------
// MI455X (gfx1250) Conformer block.
// Compute-bound (~35 GFLOP vs ~2MB live data) -> all contractions via
// v_wmma_f32_16x16x32_f16 (wave32, f32 accumulate).
//  - All GEMM operands staged as f16 -> branch-free b128/ds_b128 inner loops.
//  - GEMM W tiles and flash K/V tiles double-buffered in LDS via
//    global_load_async_to_lds_b128 (+ s_wait_asynccnt); DMA overlaps WMMA.
//  - Softmax row reductions use DPP row_ror rotation all-reduce (VALU) instead
//    of ds_bpermute.
//  - Epilogues compile-time specialized.
// ---------------------------------------------------------------------------

typedef __attribute__((ext_vector_type(16))) _Float16 v16h;
typedef __attribute__((ext_vector_type(8)))  _Float16 v8h;
typedef __attribute__((ext_vector_type(8)))  float    v8f;

union HU { v16h v; v8h p[2]; _Float16 h[16]; };
union FU { v8f  v; float f[8]; };

#define WMMA_F16(A,B,C) __builtin_amdgcn_wmma_f32_16x16x32_f16(false,(A),false,(B),(short)0,(C),false,false)

__device__ __forceinline__ int imin(int a,int b){return a<b?a:b;}
__device__ __forceinline__ int imax(int a,int b){return a>b?a:b;}

// DPP row_ror rotation within 16-lane rows: VALU-speed cross-lane reduce.
template <int CTRL>
__device__ __forceinline__ float frot(float x) {
  return __int_as_float(
      __builtin_amdgcn_mov_dpp(__float_as_int(x), CTRL, 0xf, 0xf, true));
}
__device__ __forceinline__ float rowmax16(float x) {   // max over 16-lane row
  x = fmaxf(x, frot<0x121>(x));   // row_ror:1
  x = fmaxf(x, frot<0x122>(x));   // row_ror:2
  x = fmaxf(x, frot<0x124>(x));   // row_ror:4
  x = fmaxf(x, frot<0x128>(x));   // row_ror:8
  return x;
}
__device__ __forceinline__ float rowsum16(float x) {   // sum over 16-lane row
  x += frot<0x121>(x);
  x += frot<0x122>(x);
  x += frot<0x124>(x);
  x += frot<0x128>(x);
  return x;
}

// A-operand (16x32 f16), lane l: m=l&15, hi=l>>4:
// halves[i] = row[16*hi+i]
__device__ __forceinline__ v16h load_a_f16(const _Float16* row, int hi) {
  HU u;
  u.p[0] = *(const v8h*)(row + 16 * hi);
  u.p[1] = *(const v8h*)(row + 16 * hi + 8);
  return u.v;
}

// B-operand (32x16 f16), lane l: n=l&15, hi=l>>4: halves[i] = B[16*hi+i][n]
// = 16 contiguous f16 from row n of W (32B aligned).
__device__ __forceinline__ v16h load_b_f16(const _Float16* p) {
  return *(const v16h*)p;
}

// async-stage one 64x32 f16 W tile (rows n0.., cols k0..k0+31) into LDS.
__device__ __forceinline__ void stage_w(const _Float16* W, int ldw, int n0,
                                        int k0, unsigned dstBase, int tid)
{
#pragma unroll
  for (int rep = 0; rep < 2; ++rep) {
    int id = tid + rep * 128;
    int n = id >> 2, seg = id & 3;
    const _Float16* src = W + (long long)(n0 + n) * ldw + k0 + seg * 8;
    unsigned dst = dstBase + (unsigned)(n * 64 + seg * 16);
    asm volatile("global_load_async_to_lds_b128 %0, %1, off"
                 :: "v"(dst), "v"((unsigned long long)(const void*)src)
                 : "memory");
  }
}

// ---------------------------------------------------------------------------
// GEMM: out[M,Ncols] = alpha*(A[M,K] @ W[Ncols,K]^T + bias) (+ res)
// MODE: 0 = f32 out, 1 = f32 out + residual add, 2 = f16 out.
// M, Ncols multiples of 64 (padded); K multiple of 32 -> no guards.
// Block = 128 threads (4 waves); block tile 64x64; wave tile 16x64.
// ---------------------------------------------------------------------------
template <int MODE, bool HAS_BIAS>
__global__ void __launch_bounds__(128)
gemm_wmma(const _Float16* __restrict__ A, int lda, long long aB,
          const _Float16* __restrict__ W, int ldw,
          void* __restrict__ Cv, int ldc, long long cB,
          const float* __restrict__ bias,
          const float* __restrict__ res,
          int K, float alpha)
{
  __shared__ _Float16 sW[2][64 * 32];  // double-buffered [n][k], 64B row
  const int lane = threadIdx.x & 31;
  const int m16  = lane & 15;
  const int hi   = lane >> 4;
  const int m0   = blockIdx.x * 64 + (threadIdx.x >> 5) * 16;
  const int n0   = blockIdx.y * 64;
  const long long z = blockIdx.z;
  A += z * aB;

  const unsigned base0 = (unsigned)(unsigned long long)(void*)&sW[0][0];
  const unsigned base1 = (unsigned)(unsigned long long)(void*)&sW[1][0];

  FU acc[4];
#pragma unroll
  for (int c = 0; c < 4; ++c)
#pragma unroll
    for (int i = 0; i < 8; ++i) acc[c].f[i] = 0.f;

  const _Float16* arow = A + (long long)(m0 + m16) * lda;

  stage_w(W, ldw, n0, 0, base0, threadIdx.x);
  int p = 0;
  for (int k0 = 0; k0 < K; k0 += 32) {
    asm volatile("s_wait_asynccnt 0x0" ::: "memory");
    __syncthreads();                      // tile p visible to all waves
    if (k0 + 32 < K)                      // overlap next DMA with compute
      stage_w(W, ldw, n0, k0 + 32, p ? base0 : base1, threadIdx.x);

    v16h a = load_a_f16(arow + k0, hi);
    const _Float16* sp = sW[p];
#pragma unroll
    for (int c = 0; c < 4; ++c) {
      v16h bw = load_b_f16(sp + (c * 16 + m16) * 32 + 16 * hi);
      acc[c].v = WMMA_F16(a, bw, acc[c].v);
    }
    p ^= 1;
  }

  // C layout: lane l: col = n0+c*16+(l&15); VGPR v: row = m0 + v + 8*(l>>4)
  float*    C  = (float*)Cv    + z * cB;
  _Float16* Ch = (_Float16*)Cv + z * cB;
#pragma unroll
  for (int c = 0; c < 4; ++c) {
    int cc = n0 + c * 16 + m16;
    float bv = HAS_BIAS ? bias[cc] : 0.f;
#pragma unroll
    for (int v = 0; v < 8; ++v) {
      long long idx = (long long)(m0 + v + 8 * hi) * ldc + cc;
      float val = (acc[c].f[v] + bv) * alpha;
      if constexpr (MODE == 2)      Ch[idx] = (_Float16)val;
      else if constexpr (MODE == 1) C[idx] = val + res[idx];
      else                          C[idx] = val;
    }
  }
}

// ---------------------------------------------------------------------------
// Flash attention for one batch b. Block = 4 waves, ALL of the same head h
// (16 blocks per head), covering 4 consecutive 16-row query tiles -> K and V
// tiles are shared block-wide through LDS (async-staged, double-buffered).
// q: f16 (B,1024,512), pre-scaled by 1/8. kv: f16 (B,1024,1024) [k | v].
// P: f32 (8,1024,1088): P[h,i,t], t = clip(i-j,-512,512)+512.
// out: f16 (B,1024,512).  grid = 128 blocks x 128 threads.
// ---------------------------------------------------------------------------
__global__ void __launch_bounds__(128)
flash_attn_b(const _Float16* __restrict__ q, const _Float16* __restrict__ kv,
             const float* __restrict__ P, _Float16* __restrict__ out, int b)
{
  __shared__ _Float16 sK[2][32 * 64];   // [krow][d], 128B rows
  __shared__ _Float16 sV[2][32 * 64];   // [krow][d], 128B rows
  __shared__ _Float16 sP[4][16 * 32];   // per-wave exp(S) bounce
  const int wave = threadIdx.x >> 5;
  const int lane = threadIdx.x & 31;
  const int m16  = lane & 15;
  const int hi   = lane >> 4;
  const int h    = blockIdx.x >> 4;                       // uniform per block
  const int i0   = (((blockIdx.x & 15) << 2) + wave) << 4;

  const _Float16* kvbase = kv + ((long long)b * 1024) * 1024 + h * 64;

  const _Float16* qrow = q + ((long long)(b * 1024 + i0 + m16)) * 512 + h * 64;
  v16h a0 = load_a_f16(qrow,      hi);
  v16h a1 = load_a_f16(qrow + 32, hi);

  float mrow[8], lrow[8];
  FU o[4];
#pragma unroll
  for (int v = 0; v < 8; ++v) { mrow[v] = -3.0e38f; lrow[v] = 0.f; }
#pragma unroll
  for (int c = 0; c < 4; ++c)
#pragma unroll
    for (int i = 0; i < 8; ++i) o[c].f[i] = 0.f;

  const float* Ph = P + (long long)h * 1024 * 1088;
  v8f zacc = {};

  // stage K/V rows jb..jb+31 (cols h*64..h*64+63) into LDS buffer `buf`
  auto stage_kv = [&](int jb, int buf) {
    const unsigned kb = (unsigned)(unsigned long long)(void*)&sK[buf][0];
    const unsigned vb = (unsigned)(unsigned long long)(void*)&sV[buf][0];
#pragma unroll
    for (int rep = 0; rep < 2; ++rep) {
      int id = threadIdx.x + rep * 128;      // 256 chunks of 16B each
      int n = id >> 3, seg = id & 7;
      unsigned dof = (unsigned)(n * 128 + seg * 16);
      const _Float16* srck = kvbase + (long long)(jb + n) * 1024 + seg * 8;
      asm volatile("global_load_async_to_lds_b128 %0, %1, off"
                   :: "v"(kb + dof),
                      "v"((unsigned long long)(const void*)srck) : "memory");
      asm volatile("global_load_async_to_lds_b128 %0, %1, off"
                   :: "v"(vb + dof),
                      "v"((unsigned long long)(const void*)(srck + 512))
                   : "memory");
    }
  };

  stage_kv(0, 0);
  int p = 0;
  for (int jb = 0; jb < 1024; jb += 32) {
    asm volatile("s_wait_asynccnt 0x0" ::: "memory");
    __syncthreads();                           // K/V tile p ready block-wide
    if (jb + 32 < 1024) stage_kv(jb + 32, p ^ 1);

    // --- S = Q @ K^T for the two 16-key subtiles (from LDS) ---
    const _Float16* sk = sK[p];
    FU Sa, Sb;
    Sa.v = WMMA_F16(a0, load_b_f16(sk + m16 * 64 + 16 * hi), zacc);
    Sa.v = WMMA_F16(a1, load_b_f16(sk + m16 * 64 + 32 + 16 * hi), Sa.v);
    Sb.v = WMMA_F16(a0, load_b_f16(sk + (16 + m16) * 64 + 16 * hi), zacc);
    Sb.v = WMMA_F16(a1, load_b_f16(sk + (16 + m16) * 64 + 32 + 16 * hi), Sb.v);

    // --- rel-pos add + online softmax over this 32-key block ---
#pragma unroll
    for (int v = 0; v < 8; ++v) {
      int i  = i0 + v + 8 * hi;          // C-layout row for this lane half
      int ta = imin(imax(i - (jb + m16),      -512), 512) + 512;
      int tb = imin(imax(i - (jb + 16 + m16), -512), 512) + 512;
      float sa = Sa.f[v] + Ph[(long long)i * 1088 + ta];
      float sb = Sb.f[v] + Ph[(long long)i * 1088 + tb];

      float mn   = fmaxf(mrow[v], rowmax16(fmaxf(sa, sb)));
      float corr = __expf(mrow[v] - mn);
      float pa   = __expf(sa - mn);
      float pb   = __expf(sb - mn);
      mrow[v] = mn;
      lrow[v] = lrow[v] * corr + rowsum16(pa + pb);
      o[0].f[v] *= corr; o[1].f[v] *= corr; o[2].f[v] *= corr; o[3].f[v] *= corr;
      sP[wave][(v + 8 * hi) * 32 + m16]      = (_Float16)pa;
      sP[wave][(v + 8 * hi) * 32 + 16 + m16] = (_Float16)pb;
    }
    // sP is per-wave; DS ops are in-order within a wave -> compiler fence only
    __builtin_amdgcn_wave_barrier();

    HU pu;   // reload exp(S) tile in A-operand layout
    pu.p[0] = *(const v8h*)&sP[wave][m16 * 32 + 16 * hi];
    pu.p[1] = *(const v8h*)&sP[wave][m16 * 32 + 16 * hi + 8];

    // --- O += P(16x32) @ V(32x64), V gathered from LDS ---
    const _Float16* sv = sV[p];
#pragma unroll
    for (int c = 0; c < 4; ++c) {
      HU u;
#pragma unroll
      for (int i = 0; i < 16; ++i)
        u.h[i] = sv[(16 * hi + i) * 64 + c * 16 + m16];
      o[c].v = WMMA_F16(pu.v, u.v, o[c].v);
    }
    __builtin_amdgcn_wave_barrier();   // keep next stores after these loads
    p ^= 1;
  }

#pragma unroll
  for (int v = 0; v < 8; ++v) {
    int i = i0 + v + 8 * hi;
    float inv = 1.f / lrow[v];
#pragma unroll
    for (int c = 0; c < 4; ++c)
      out[((long long)(b * 1024 + i)) * 512 + h * 64 + c * 16 + m16] =
          (_Float16)(o[c].f[v] * inv);
  }
}

// ---------------------------------------------------------------------------
// LayerNorm over last dim (64). One wave per row; T = float or _Float16.
// ---------------------------------------------------------------------------
template <typename T>
__global__ void __launch_bounds__(256)
layernorm64(const float* __restrict__ x, const float* __restrict__ g,
            const float* __restrict__ bb, T* __restrict__ y, int rows)
{
  int row  = (blockIdx.x * blockDim.x + threadIdx.x) >> 5;
  int lane = threadIdx.x & 31;
  if (row >= rows) return;
  const float* xr = x + (long long)row * 64;
  float2 v = ((const float2*)xr)[lane];
  float s = v.x + v.y, ss = v.x * v.x + v.y * v.y;
#pragma unroll
  for (int off = 16; off > 0; off >>= 1) {
    s  += __shfl_xor(s,  off, 32);
    ss += __shfl_xor(ss, off, 32);
  }
  float mean = s * (1.f / 64.f);
  float var  = ss * (1.f / 64.f) - mean * mean;
  float inv  = rsqrtf(var + 1e-5f);
  T* yr = y + (long long)row * 64;
  int c = lane * 2;
  yr[c]     = (T)((v.x - mean) * inv * g[c]     + bb[c]);
  yr[c + 1] = (T)((v.y - mean) * inv * g[c + 1] + bb[c + 1]);
}

// ---------------------------------------------------------------------------
// Packing / activation kernels
// ---------------------------------------------------------------------------
__global__ void cvt_f16(const float* __restrict__ s, _Float16* __restrict__ d)
{
  int i = blockIdx.x * 256 + threadIdx.x;
  d[i] = (_Float16)s[i];
}

// rel_emb (1025x64) -> zero-padded f16 (1088x64)
__global__ void rel_pad(const float* __restrict__ s, _Float16* __restrict__ d)
{
  int i = blockIdx.x * 256 + threadIdx.x;              // 1088*64
  d[i] = (i >> 6) < 1025 ? (_Float16)s[i] : (_Float16)0.f;
}

// im1[b*1027+t'][ic*4+kk] = ln[b, t'+kk-3, ic]; rows 8216..8255 zero pad
__global__ void im2col_ff1(const _Float16* __restrict__ ln, _Float16* __restrict__ im1)
{
  int idx = blockIdx.x * 256 + threadIdx.x;            // 8256*256
  int r = idx >> 8, c = idx & 255;
  _Float16 v = (_Float16)0.f;
  if (r < 8216) {
    int b = r / 1027, tp = r - b * 1027;
    int t = tp + (c & 3) - 3;
    if (t >= 0 && t < 1024) v = ln[((long long)((b << 10) + t)) * 64 + (c >> 2)];
  }
  im1[idx] = v;
}

// mid = a * silu(g) from conv1 output h1 (8256x512 f32) -> f16 (8216x256)
__global__ void silu_mid(const float* __restrict__ h1, _Float16* __restrict__ mid)
{
  int idx = blockIdx.x * 256 + threadIdx.x;            // 8216*256
  int r = idx >> 8, c = idx & 255;
  float a = h1[(long long)r * 512 + c];
  float g = h1[(long long)r * 512 + 256 + c];
  mid[idx] = (_Float16)(a * g / (1.f + __expf(-g)));
}

// im2[b*1024+t][c*4+k] = mid[b*1027 + t + k][c]
__global__ void im2col_ff2(const _Float16* __restrict__ mid, _Float16* __restrict__ im2)
{
  int idx = blockIdx.x * 256 + threadIdx.x;            // 8192*1024
  int r = idx >> 10, c2 = idx & 1023;
  int b = r >> 10, t = r & 1023;
  im2[idx] = mid[(long long)(b * 1027 + t + (c2 & 3)) * 256 + (c2 >> 2)];
}

// W2[oc][c*4+k] = dw[c, oc, 3-k]   (dw: (256,64,4) f32) -> f16
__global__ void w2_prep(const float* __restrict__ dw, _Float16* __restrict__ w2)
{
  int idx = blockIdx.x * 256 + threadIdx.x;            // 64*1024
  int oc = idx >> 10, c2 = idx & 1023;
  int c = c2 >> 2, k = c2 & 3;
  w2[idx] = (_Float16)dw[((long long)(c * 64 + oc)) * 4 + (3 - k)];
}

__global__ void glu_gate(const float* __restrict__ pw1, float* __restrict__ glu)
{
  int idx = blockIdx.x * 256 + threadIdx.x;            // 8192*128
  int r = idx >> 7, ch = idx & 127;
  float a = pw1[(long long)r * 256 + ch];
  float g = pw1[(long long)r * 256 + 128 + ch];
  glu[idx] = a / (1.f + __expf(-g));
}

// depthwise conv K=31 pad 15; layout (b, t, ch), ch contiguous
__global__ void dwconv31(const float* __restrict__ x, const float* __restrict__ w,
                         const float* __restrict__ bias, float* __restrict__ y)
{
  int idx = blockIdx.x * 256 + threadIdx.x;            // 8*1024*128
  int ch = idx & 127, t = (idx >> 7) & 1023, b = idx >> 17;
  float acc = bias[ch];
#pragma unroll
  for (int k = 0; k < 31; ++k) {
    int tt = t + k - 15;
    if (tt >= 0 && tt < 1024)
      acc += w[ch * 31 + k] * x[((long long)((b << 10) + tt)) * 128 + ch];
  }
  y[idx] = acc;
}

__global__ void bn_stats(const float* __restrict__ h, float* __restrict__ st)
{
  int ch = blockIdx.x, tid = threadIdx.x;
  float s = 0.f, ss = 0.f;
  for (int i = tid; i < 8192; i += 256) {
    float v = h[(long long)i * 128 + ch];
    s += v; ss += v * v;
  }
  __shared__ float sb[256], sb2[256];
  sb[tid] = s; sb2[tid] = ss;
  __syncthreads();
  for (int o = 128; o > 0; o >>= 1) {
    if (tid < o) { sb[tid] += sb[tid + o]; sb2[tid] += sb2[tid + o]; }
    __syncthreads();
  }
  if (tid == 0) {
    float m = sb[0] * (1.f / 8192.f);
    st[ch] = m;
    st[128 + ch] = sb2[0] * (1.f / 8192.f) - m * m;
  }
}

// BN + swish, f32 in -> f16 out (A-matrix of pw2 GEMM)
__global__ void bn_swish(const float* __restrict__ h, const float* __restrict__ g,
                         const float* __restrict__ b, const float* __restrict__ st,
                         _Float16* __restrict__ y)
{
  int idx = blockIdx.x * 256 + threadIdx.x;            // 8192*128
  int ch = idx & 127;
  float nm = (h[idx] - st[ch]) * rsqrtf(st[128 + ch] + 1e-5f) * g[ch] + b[ch];
  y[idx] = (_Float16)(nm / (1.f + __expf(-nm)));
}

// ---------------------------------------------------------------------------
// Host side.  Offsets in floats (f16 buffers: 2 halves per float).
// ---------------------------------------------------------------------------
static const long long OFF_XRES = 0;             // f32 8192x64
static const long long OFF_LN16 = 524288;        // f16 8192x64  (262144 fl)
static const long long A0       = 786432;        // stage-overlapped arena
// FF stage
static const long long F_IM1 = A0;               // f16 8256x256 (1056768 fl)
static const long long F_H1  = F_IM1 + 1056768;  // f32 8256x512 (4227072 fl)
static const long long F_MID = F_H1  + 4227072;  // f16 8216x256 (1051648 fl)
static const long long F_IM2 = F_MID + 1051648;  // f16 8192x1024 (4194304 fl)
static const long long F_W1  = F_IM2 + 4194304;  // f16 512x256  (65536 fl)
static const long long F_W2  = F_W1  + 65536;    // f16 64x1024  (32768 fl)
// Attention stage
static const long long T_Q   = A0;               // f16 8192x512 (2097152 fl)
static const long long T_KV  = T_Q   + 2097152;  // f16 8192x1024 (4194304 fl)
static const long long T_AO  = T_KV  + 4194304;  // f16 8192x512 (2097152 fl)
static const long long T_P   = T_AO  + 2097152;  // f32 8x1024x1088 (8912896 fl)
static const long long T_WQ  = T_P   + 8912896;  // f16 512x64
static const long long T_WKV = T_WQ  + 16384;    // f16 1024x64
static const long long T_WO  = T_WKV + 32768;    // f16 64x512
static const long long T_REL = T_WO  + 16384;    // f16 1088x64 (34816 fl)
// Conv stage
static const long long C_PW1 = A0;               // f32 8192x256
static const long long C_GLU = C_PW1 + 2097152;  // f32 8192x128
static const long long C_DW  = C_GLU + 1048576;  // f32 8192x128
static const long long C_D16 = C_DW  + 1048576;  // f16 8192x128 (524288 fl)
static const long long C_ST  = C_D16 + 524288;   // f32 256
static const long long C_W1  = C_ST  + 256;      // f16 256x64 (8192 fl)
static const long long C_W2  = C_W1  + 8192;     // f16 64x128 (4096 fl)

static void ff_stage(float* ws, const float* lng, const float* lnb,
                     const float* cw, const float* cb,
                     const float* dwp, const float* dbp, hipStream_t s)
{
  float* xres = ws + OFF_XRES;
  _Float16* ln16 = (_Float16*)(ws + OFF_LN16);
  _Float16* im1  = (_Float16*)(ws + F_IM1);
  float*    h1   = ws + F_H1;
  _Float16* mid  = (_Float16*)(ws + F_MID);
  _Float16* im2  = (_Float16*)(ws + F_IM2);
  _Float16* w1h  = (_Float16*)(ws + F_W1);
  _Float16* w2h  = (_Float16*)(ws + F_W2);

  layernorm64<_Float16><<<1024, 256, 0, s>>>(xres, lng, lnb, ln16, 8192);
  cvt_f16<<<512, 256, 0, s>>>(cw, w1h);                     // 512*256
  w2_prep<<<256, 256, 0, s>>>(dwp, w2h);                    // 64*1024
  im2col_ff1<<<8256, 256, 0, s>>>(ln16, im1);
  // h1(8256x512) = im1(8256x256) @ cw(512x256)^T + cb
  gemm_wmma<0, true><<<dim3(129, 8, 1), 128, 0, s>>>(im1, 256, 0, w1h, 256,
      h1, 512, 0, cb, nullptr, 256, 1.0f);
  silu_mid<<<8216, 256, 0, s>>>(h1, mid);
  im2col_ff2<<<32768, 256, 0, s>>>(mid, im2);
  // xres += 0.5*(im2(8192x1024) @ W2(64x1024)^T + db)
  gemm_wmma<1, true><<<dim3(128, 1, 1), 128, 0, s>>>(im2, 1024, 0, w2h, 1024,
      xres, 64, 0, dbp, xres, 1024, 0.5f);
}

extern "C" void kernel_launch(void* const* d_in, const int* in_sizes, int n_in,
                              void* d_out, int out_size, void* d_ws, size_t ws_size,
                              hipStream_t stream)
{
  (void)in_sizes; (void)n_in; (void)out_size; (void)ws_size;
  const float* x        = (const float*)d_in[0];
  const float* ff1_ln_g = (const float*)d_in[1];
  const float* ff1_ln_b = (const float*)d_in[2];
  const float* ff1_cw   = (const float*)d_in[3];
  const float* ff1_cb   = (const float*)d_in[4];
  const float* ff1_dw   = (const float*)d_in[5];
  const float* ff1_db   = (const float*)d_in[6];
  const float* attn_lng = (const float*)d_in[7];
  const float* attn_lnb = (const float*)d_in[8];
  const float* wq       = (const float*)d_in[9];
  const float* wkv      = (const float*)d_in[10];
  const float* wo       = (const float*)d_in[11];
  const float* wo_b     = (const float*)d_in[12];
  const float* rel_emb  = (const float*)d_in[13];
  const float* conv_lng = (const float*)d_in[14];
  const float* conv_lnb = (const float*)d_in[15];
  const float* pw1_w    = (const float*)d_in[16];
  const float* pw1_b    = (const float*)d_in[17];
  const float* dw_w     = (const float*)d_in[18];
  const float* dw_b     = (const float*)d_in[19];
  const float* bn_g     = (const float*)d_in[20];
  const float* bn_b     = (const float*)d_in[21];
  const float* pw2_w    = (const float*)d_in[22];
  const float* pw2_b    = (const float*)d_in[23];
  const float* ff2_ln_g = (const float*)d_in[24];
  const float* ff2_ln_b = (const float*)d_in[25];
  const float* ff2_cw   = (const float*)d_in[26];
  const float* ff2_cb   = (const float*)d_in[27];
  const float* ff2_dw   = (const float*)d_in[28];
  const float* ff2_db   = (const float*)d_in[29];
  const float* post_g   = (const float*)d_in[30];
  const float* post_b   = (const float*)d_in[31];

  float* ws = (float*)d_ws;
  float* xres = ws + OFF_XRES;
  _Float16* ln16 = (_Float16*)(ws + OFF_LN16);

  hipMemcpyAsync(xres, x, (size_t)524288 * sizeof(float),
                 hipMemcpyDeviceToDevice, stream);

  // ---- FF1 ----
  ff_stage(ws, ff1_ln_g, ff1_ln_b, ff1_cw, ff1_cb, ff1_dw, ff1_db, stream);

  // ---- Attention ----
  {
    _Float16* qh   = (_Float16*)(ws + T_Q);
    _Float16* kvh  = (_Float16*)(ws + T_KV);
    _Float16* aoh  = (_Float16*)(ws + T_AO);
    float*    Pf   = ws + T_P;
    _Float16* wqh  = (_Float16*)(ws + T_WQ);
    _Float16* wkvh = (_Float16*)(ws + T_WKV);
    _Float16* woh  = (_Float16*)(ws + T_WO);
    _Float16* relh = (_Float16*)(ws + T_REL);

    layernorm64<_Float16><<<1024, 256, 0, stream>>>(xres, attn_lng, attn_lnb,
                                                    ln16, 8192);
    cvt_f16<<<128, 256, 0, stream>>>(wq, wqh);              // 512*64
    cvt_f16<<<256, 256, 0, stream>>>(wkv, wkvh);            // 1024*64
    cvt_f16<<<128, 256, 0, stream>>>(wo, woh);              // 64*512
    rel_pad<<<272, 256, 0, stream>>>(rel_emb, relh);        // 1088*64

    // q = 0.125 * ln @ wq^T (scale folds into both score terms), f16 out
    gemm_wmma<2, false><<<dim3(128, 8, 1), 128, 0, stream>>>(ln16, 64, 0,
        wqh, 64, qh, 512, 0, nullptr, nullptr, 64, 0.125f);
    // kv = ln @ wkv^T, f16 out
    gemm_wmma<2, false><<<dim3(128, 16, 1), 128, 0, stream>>>(ln16, 64, 0,
        wkvh, 64, kvh, 1024, 0, nullptr, nullptr, 64, 1.0f);

    for (int b = 0; b < 8; ++b) {
      // P[h,i,m] = q[b,h,i,:] . rel_emb[m,:]  (heads batched via z)
      gemm_wmma<0, false><<<dim3(16, 17, 8), 128, 0, stream>>>(
          qh + (long long)b * 1024 * 512, 512, /*aB=*/64,
          relh, 64,
          Pf, 1088, /*cB=*/(long long)1024 * 1088,
          nullptr, nullptr, 64, 1.0f);
      flash_attn_b<<<128, 128, 0, stream>>>(qh, kvh, Pf, aoh, b);
    }
    // xres += attnout @ wo^T + wo_b
    gemm_wmma<1, true><<<dim3(128, 1, 1), 128, 0, stream>>>(aoh, 512, 0,
        woh, 512, xres, 64, 0, wo_b, xres, 512, 1.0f);
  }

  // ---- Conv module ----
  {
    float*    pw1o = ws + C_PW1;
    float*    glu  = ws + C_GLU;
    float*    dwo  = ws + C_DW;
    _Float16* d16  = (_Float16*)(ws + C_D16);
    float*    st   = ws + C_ST;
    _Float16* w1h  = (_Float16*)(ws + C_W1);
    _Float16* w2h  = (_Float16*)(ws + C_W2);

    layernorm64<_Float16><<<1024, 256, 0, stream>>>(xres, conv_lng, conv_lnb,
                                                    ln16, 8192);
    cvt_f16<<<64, 256, 0, stream>>>(pw1_w, w1h);            // 256*64
    cvt_f16<<<32, 256, 0, stream>>>(pw2_w, w2h);            // 64*128
    gemm_wmma<0, true><<<dim3(128, 4, 1), 128, 0, stream>>>(ln16, 64, 0,
        w1h, 64, pw1o, 256, 0, pw1_b, nullptr, 64, 1.0f);
    glu_gate<<<4096, 256, 0, stream>>>(pw1o, glu);
    dwconv31<<<4096, 256, 0, stream>>>(glu, dw_w, dw_b, dwo);
    bn_stats<<<128, 256, 0, stream>>>(dwo, st);
    bn_swish<<<4096, 256, 0, stream>>>(dwo, bn_g, bn_b, st, d16);
    gemm_wmma<1, true><<<dim3(128, 1, 1), 128, 0, stream>>>(d16, 128, 0,
        w2h, 128, xres, 64, 0, pw2_b, xres, 128, 1.0f);
  }

  // ---- FF2 ----
  ff_stage(ws, ff2_ln_g, ff2_ln_b, ff2_cw, ff2_cb, ff2_dw, ff2_db, stream);

  // ---- Final LayerNorm -> d_out (f32) ----
  layernorm64<float><<<1024, 256, 0, stream>>>(xres, post_g, post_b,
                                               (float*)d_out, 8192);
}
